// SubQuadraticHeterogenousCrossAttention_23038204576030
// MI455X (gfx1250) — matrix operand order repaired and checked by reading, not compiled, not run on hardware
//
#include <hip/hip_runtime.h>
#include <hip/hip_bf16.h>
#include <stdint.h>

// ---------------------------------------------------------------------------
// SubQuadraticHeterogenousCrossAttention for MI455X (gfx1250).
// bf16 WMMA GEMMs + async global->LDS (ASYNCcnt) pipeline + v_permlane16
// lane reductions (no ds_bpermute / dscnt stalls).
// Head h == timestep t; softmax/rope act on 16-channel blocks of linear rows;
// x_0 branch is dead (reference overwrites `out`).
// ---------------------------------------------------------------------------

typedef __bf16 bf16;
typedef __attribute__((ext_vector_type(16))) __bf16 v16bf;
typedef __attribute__((ext_vector_type(8)))  __bf16 v8bf;
typedef __attribute__((ext_vector_type(8)))  float  v8f;
typedef __attribute__((ext_vector_type(4)))  float  v4f;

#define DEV __device__ __forceinline__

static constexpr float LOG1E4_OVER8 = 1.1512925465f; // ln(10000)/8

// permlane16 xor-selector nibble tables (lane i reads lane i^m within its 16)
#define SELX1_LO 0x67452301u
#define SELX1_HI 0xEFCDAB89u
#define SELX2_LO 0x54761032u
#define SELX2_HI 0xDCFE98BAu
#define SELX4_LO 0x32107654u
#define SELX4_HI 0xBA98FEDCu
#define SELX8_LO 0xFEDCBA98u
#define SELX8_HI 0x76543210u
#define SELID_LO 0x76543210u
#define SELID_HI 0xFEDCBA98u

DEV float pl16(float v, unsigned lo, unsigned hi) {
#if __has_builtin(__builtin_amdgcn_permlane16)
  unsigned u = __float_as_uint(v);
  return __uint_as_float(__builtin_amdgcn_permlane16(u, u, lo, hi, false, false));
#else
  int m = (lo == SELX1_LO) ? 1 : (lo == SELX2_LO) ? 2 : (lo == SELX4_LO) ? 4 : 8;
  return __shfl_xor(v, m, 32);
#endif
}
DEV float pswap16(float v) {  // lane <-> lane^16
#if __has_builtin(__builtin_amdgcn_permlanex16)
  unsigned u = __float_as_uint(v);
  return __uint_as_float(
      __builtin_amdgcn_permlanex16(u, u, SELID_LO, SELID_HI, false, false));
#else
  return __shfl_xor(v, 16, 32);
#endif
}

DEV float redmax16(float v) {
  v = fmaxf(v, pl16(v, SELX1_LO, SELX1_HI));
  v = fmaxf(v, pl16(v, SELX2_LO, SELX2_HI));
  v = fmaxf(v, pl16(v, SELX4_LO, SELX4_HI));
  v = fmaxf(v, pl16(v, SELX8_LO, SELX8_HI));
  return v;
}
DEV float redsum16(float v) {
  v += pl16(v, SELX1_LO, SELX1_HI);
  v += pl16(v, SELX2_LO, SELX2_HI);
  v += pl16(v, SELX4_LO, SELX4_HI);
  v += pl16(v, SELX8_LO, SELX8_HI);
  return v;
}

DEV v16bf cat8(v8bf a, v8bf b) {
  return __builtin_shufflevector(a, b, 0,1,2,3,4,5,6,7,8,9,10,11,12,13,14,15);
}

// A fragment (16xK row-major bf16 tile, row stride `stride` halves), k-step kt.
DEV v16bf frag_a(const bf16* t, int stride, int kt, int lane) {
  int m = lane & 15, hb = (lane >> 4) & 1;
  const bf16* p = t + m * stride + kt * 32 + 16 * hb;
  return cat8(*(const v8bf*)p, *(const v8bf*)(p + 8));
}

// B fragment from row-major weight W[o][i] (stride 128): B[k=i, n=o].
DEV v16bf frag_b_w(const bf16* w, int ct, int kt, int lane) {
  int n = lane & 15, hb = (lane >> 4) & 1;
  return *(const v16bf*)(w + (ct * 16 + n) * 128 + kt * 32 + 16 * hb);
}

// B fragment from LDS tile stored [n][k], row stride `stride` halves.
DEV v16bf frag_b_t(const bf16* t, int stride, int kt, int lane) {
  int n = lane & 15, hb = (lane >> 4) & 1;
  const bf16* p = t + n * stride + kt * 32 + 16 * hb;
  return cat8(*(const v8bf*)p, *(const v8bf*)(p + 8));
}

DEV v8f wmma_bf16(v16bf a, v16bf b, v8f c) {
  return __builtin_amdgcn_wmma_f32_16x16x32_bf16(false, a, false, b, (short)0, c,
                                                 false, false);
}

// Async DMA: global (16B/lane) -> LDS, tracked by ASYNCcnt.
DEV void async_b128(unsigned lds_addr, const void* gaddr) {
  asm volatile("global_load_async_to_lds_b128 %0, %1, off"
               :: "v"(lds_addr), "v"((unsigned long long)(uintptr_t)gaddr)
               : "memory");
}
DEV void wait_async0() { asm volatile("s_wait_asynccnt 0x0" ::: "memory"); }
DEV unsigned lds_off(const void* p) { return (unsigned)(uintptr_t)p; }

// ---------------------------------------------------------------------------
// Kernel P: weights f32 -> bf16 (row-major (out,in)) + zero accumulators.
// ---------------------------------------------------------------------------
__global__ void __launch_bounds__(256) prep_kernel(
    const float* __restrict__ Wk, const float* __restrict__ Wv,
    const float* __restrict__ Wq, const float* __restrict__ Wo,
    bf16* __restrict__ WkBf, bf16* __restrict__ WvBf,
    bf16* __restrict__ WqBf, bf16* __restrict__ WoBf,
    float* __restrict__ accum) {
  int t = blockIdx.x * 256 + threadIdx.x;   // 0..65535
  int i = t & 16383;
  int w = t >> 14;
  if (w == 0)      WkBf[i] = (bf16)Wk[i];
  else if (w == 1) WvBf[i] = (bf16)Wv[i];
  else if (w == 2) WqBf[i] = (bf16)Wq[i];
  else             WoBf[i] = (bf16)Wo[i];
  if (t < 8704) accum[t] = 0.f;             // kv_acc (8192) + ksum_acc (512)
}

// ---------------------------------------------------------------------------
// Kernel 1: edge branch, persistent over 8 chunks with async double-buffering.
// Per (bt, chunk): k = softmax(rope(X@Wk^T+bk)), v = X@Wv^T+bv; accumulate
// kv[16x16] = k^T v (in registers across chunks) and ksum[16] per bt.
// LDS: Xf32 64K | Xbf 32K | Kt 34816 | Vt 34816 | kvred 8K | ksred 512
//      = 176640 B
// ---------------------------------------------------------------------------
__global__ void __launch_bounds__(256) kv_reduce_kernel(
    const float* __restrict__ edge, const float* __restrict__ bk,
    const float* __restrict__ bv, const bf16* __restrict__ WkBf,
    const bf16* __restrict__ WvBf, float* __restrict__ kv_acc,
    float* __restrict__ ksum_acc) {
  extern __shared__ __align__(16) char smem[];
  const int tid = threadIdx.x, lane = tid & 31, wave = tid >> 5;
  const int hb = (lane >> 4) & 1;
  const int bt = blockIdx.y;
  const int cset = blockIdx.x;  // 8 chunksets of 8x128 rows

  float* Xf    = (float*)smem + wave * (16 * 128);
  bf16*  Xbf   = (bf16*)(smem + 65536) + wave * (16 * 128);
  bf16*  Kt    = (bf16*)(smem + 98304) + wave * (16 * 136);
  bf16*  Vt    = (bf16*)(smem + 133120) + wave * (16 * 136);
  float* kvred = (float*)(smem + 167936) + wave * 256;
  float* ksred = (float*)(smem + 176128) + wave * 16;

  if (tid < 128) {  // pull bf16 weights toward L2 early
    __builtin_prefetch(WkBf + tid * 128, 0, 1);
    __builtin_prefetch(WvBf + tid * 128, 0, 1);
  }

  const float*   gbase  = edge + (size_t)bt * 8192 * 128;
  const unsigned xfBase = lds_off(Xf);

  auto issueAsync = [&](int c) {
    const float* src = gbase + (size_t)((cset * 8 + c) * 128 + wave * 16) * 128;
    #pragma unroll
    for (int it = 0; it < 16; ++it)
      async_b128(xfBase + (unsigned)((it * 128 + lane * 4) * 4),
                 src + it * 128 + lane * 4);
  };
  auto waitCvt = [&]() {   // ASYNCcnt drain + f32 -> bf16 tile
    wait_async0();
    #pragma unroll
    for (int it = 0; it < 16; ++it) {
      v4f x = *(const v4f*)(Xf + it * 128 + lane * 4);
      bf16* d = Xbf + it * 128 + lane * 4;
      d[0] = (bf16)x.x; d[1] = (bf16)x.y; d[2] = (bf16)x.z; d[3] = (bf16)x.w;
    }
  };

  issueAsync(0);
  waitCvt();

  const int   p    = (lane & 15) >> 1;
  const float freq = __expf(-(float)p * LOG1E4_OVER8);

  // chunk-invariant biases, hoisted out of all loops
  float bkr[8], bvr[8];
  #pragma unroll
  for (int ct = 0; ct < 8; ++ct) {
    bkr[ct] = bk[ct * 16 + (lane & 15)];
    bvr[ct] = bv[ct * 16 + (lane & 15)];
  }

  v8f   kvp = {};
  float ksl = 0.f;

  for (int c = 0; c < 8; ++c) {
    v16bf afr[4];
    #pragma unroll
    for (int kt = 0; kt < 4; ++kt) afr[kt] = frag_a(Xbf, 128, kt, lane);

    if (c < 7) issueAsync(c + 1);  // DMA next chunk while WMMAs run

    const int m0 = (cset * 8 + c) * 128 + wave * 16;
    float cs[8], sn[8];
    #pragma unroll
    for (int g = 0; g < 8; ++g) {
      int pos = (m0 + g + 8 * hb) >> 10;            // k rope: pos = m/1024
      __sincosf((float)pos * freq, &sn[g], &cs[g]);
    }

    #pragma unroll
    for (int ct = 0; ct < 8; ++ct) {
      // ---- k column tile: GEMM, +bias, rope, softmax across 16 lanes ----
      v8f ak = {};
      #pragma unroll
      for (int kt = 0; kt < 4; ++kt)
        ak = wmma_bf16(afr[kt], frag_b_w(WkBf, ct, kt, lane), ak);
      #pragma unroll
      for (int g = 0; g < 8; ++g) {
        float v = ak[g] + bkr[ct];
        float partner = pl16(v, SELX1_LO, SELX1_HI);            // rope pair
        v = v * cs[g] + partner * ((lane & 1) ? sn[g] : -sn[g]);
        float mx = redmax16(v);
        float e  = __expf(v - mx);
        float s  = redsum16(e);
        v = e / s;
        ksl += v;
        Kt[(lane & 15) * 136 + ct * 16 + g + 8 * hb] = (bf16)v;  // Kt[d][s]
      }
      // ---- v column tile: GEMM + bias, transposed store Vt[e][s] ----
      v8f av = {};
      #pragma unroll
      for (int kt = 0; kt < 4; ++kt)
        av = wmma_bf16(afr[kt], frag_b_w(WvBf, ct, kt, lane), av);
      #pragma unroll
      for (int g = 0; g < 8; ++g)
        Vt[(lane & 15) * 136 + ct * 16 + g + 8 * hb] = (bf16)(av[g] + bvr[ct]);
    }

    // kv += k^T v over this chunk's 128 s-rows (register accumulator).
    #pragma unroll
    for (int kt = 0; kt < 4; ++kt)
      kvp = wmma_bf16(frag_a(Kt, 136, kt, lane), frag_b_t(Vt, 136, kt, lane),
                      kvp);

    if (c < 7) waitCvt();  // next chunk's tile becomes ready
  }

  ksl += pswap16(ksl);  // fold the two M-halves

  #pragma unroll
  for (int g = 0; g < 8; ++g)
    kvred[(g + 8 * hb) * 16 + (lane & 15)] = kvp[g];
  if (lane < 16) ksred[lane] = ksl;
  __syncthreads();

  // Cross-wave reduce, then one global f32 atomic per element.
  float s = 0.f;
  #pragma unroll
  for (int w = 0; w < 8; ++w) s += ((float*)(smem + 167936))[w * 256 + tid];
  atomicAdd(&kv_acc[bt * 256 + tid], s);
  if (tid < 16) {
    float t = 0.f;
    #pragma unroll
    for (int w = 0; w < 8; ++w) t += ((float*)(smem + 176128))[w * 16 + tid];
    atomicAdd(&ksum_acc[bt * 16 + tid], t);
  }
}

// ---------------------------------------------------------------------------
// Kernel 2: q = rope(softmax(X@Wq^T+bq)); out_h = q + (q@kv)*Dinv;
// gather heads -> PreWo(128x128) -> @Wo^T + bo -> scattered final store.
// LDS: Xf32 64K | Xbf 32K | Qblk 10240 | kvT 8192 | PreWo 34816 = 151552 B
// ---------------------------------------------------------------------------
__global__ void __launch_bounds__(256) out_kernel(
    const float* __restrict__ qdat, const float* __restrict__ bq,
    const float* __restrict__ bo, const bf16* __restrict__ WqBf,
    const bf16* __restrict__ WoBf, const float* __restrict__ kv_acc,
    const float* __restrict__ ksum_acc, float* __restrict__ out) {
  extern __shared__ __align__(16) char smem[];
  const int tid = threadIdx.x, lane = tid & 31, wave = tid >> 5;
  const int hb = (lane >> 4) & 1;
  const int b  = blockIdx.y;
  const int n0 = blockIdx.x * 16;
  const int h  = wave;
  const int bt = b * 8 + h;

  float* Xf    = (float*)smem + wave * (16 * 128);
  bf16*  Xbf   = (bf16*)(smem + 65536) + wave * (16 * 128);
  bf16*  Qblk  = (bf16*)(smem + 98304) + wave * (16 * 40);   // K padded w/ 0
  bf16*  kvT   = (bf16*)(smem + 108544) + wave * (16 * 32);  // [e][k], K>=16 = 0
  bf16*  PreWo = (bf16*)(smem + 116736);                     // 128 x 136 halves

  // Kick the q-tile DMA first; overlap setup below with it.
  {
    const float*   src    = qdat + (size_t)(bt * 2048 + n0) * 128;
    const unsigned xfBase = lds_off(Xf);
    #pragma unroll
    for (int it = 0; it < 16; ++it)
      async_b128(xfBase + (unsigned)((it * 128 + lane * 4) * 4),
                 src + it * 128 + lane * 4);
  }

  if (tid < 128) {
    __builtin_prefetch(WqBf + tid * 128, 0, 1);
    __builtin_prefetch(WoBf + tid * 128, 0, 1);
  }
  for (int i = lane; i < 16 * 40; i += 32) Qblk[i] = (bf16)0.f;
  for (int i = lane; i < 16 * 32; i += 32) kvT[i]  = (bf16)0.f;
  for (int i = lane; i < 256; i += 32) {                  // kv[d][e] -> kvT[e][d]
    int d = i >> 4, e = i & 15;
    kvT[e * 32 + d] = (bf16)kv_acc[bt * 256 + i];
  }
  const float ksv = ksum_acc[bt * 16 + (lane & 15)];
  float bqr[8];
  #pragma unroll
  for (int ct = 0; ct < 8; ++ct) bqr[ct] = bq[ct * 16 + (lane & 15)];

  wait_async0();
  #pragma unroll
  for (int it = 0; it < 16; ++it) {
    v4f x = *(const v4f*)(Xf + it * 128 + lane * 4);
    bf16* d = Xbf + it * 128 + lane * 4;
    d[0] = (bf16)x.x; d[1] = (bf16)x.y; d[2] = (bf16)x.z; d[3] = (bf16)x.w;
  }

  v16bf afr[4];
  #pragma unroll
  for (int kt = 0; kt < 4; ++kt) afr[kt] = frag_a(Xbf, 128, kt, lane);

  const int   p    = (lane & 15) >> 1;
  const float freq = __expf(-(float)p * LOG1E4_OVER8);
  float cs[8], sn[8];
  #pragma unroll
  for (int g = 0; g < 8; ++g) {
    int pos = (n0 + g + 8 * hb) >> 8;  // q rope: pos = n/256
    __sincosf((float)pos * freq, &sn[g], &cs[g]);
  }

  #pragma unroll
  for (int ct = 0; ct < 8; ++ct) {
    v8f aq = {};
    #pragma unroll
    for (int kt = 0; kt < 4; ++kt)
      aq = wmma_bf16(afr[kt], frag_b_w(WqBf, ct, kt, lane), aq);
    float qreg[8], dinv[8];
    #pragma unroll
    for (int g = 0; g < 8; ++g) {
      float v = aq[g] + bqr[ct];
      // softmax FIRST (q order: softmax then rope)
      float mx = redmax16(v);
      float e  = __expf(v - mx);
      float s  = redsum16(e);
      v = e / s;
      // rope
      float partner = pl16(v, SELX1_LO, SELX1_HI);
      v = v * cs[g] + partner * ((lane & 1) ? sn[g] : -sn[g]);
      // Dinv per (row, block): 1 / max(sum_c q*ksum, 1e-8)
      float t = redsum16(v * ksv);
      dinv[g] = 1.f / fmaxf(t, 1e-8f);
      qreg[g] = v;
      Qblk[(g + 8 * hb) * 40 + (lane & 15)] = (bf16)v;  // A[M=row][K=d]
    }
    // q(16x16) @ kv(16x16) via one padded-K bf16 WMMA.
    v8f zero = {};
    v8f dkv  = wmma_bf16(frag_a(Qblk, 40, 0, lane),
                         frag_b_t(kvT, 32, 0, lane), zero);
    #pragma unroll
    for (int g = 0; g < 8; ++g) {
      float o = qreg[g] + dkv[g] * dinv[g];
      // PreWo row = r*8 + j (r = node offset = M, j = ct), col = h*16 + c'
      PreWo[((g + 8 * hb) * 8 + ct) * 136 + h * 16 + (lane & 15)] = (bf16)o;
    }
  }
  __syncthreads();

  // Final GEMM: wave = 16-row tile of PreWo; out = PreWo @ Wo^T + bo.
  const bf16* Arow = PreWo + wave * 16 * 136;
  v16bf afo[4];
  #pragma unroll
  for (int kt = 0; kt < 4; ++kt) afo[kt] = frag_a(Arow, 136, kt, lane);
  #pragma unroll
  for (int ct = 0; ct < 8; ++ct) {
    v8f ao = {};
    #pragma unroll
    for (int kt = 0; kt < 4; ++kt)
      ao = wmma_bf16(afo[kt], frag_b_w(WoBf, ct, kt, lane), ao);
    const float bov = bo[ct * 16 + (lane & 15)];
    #pragma unroll
    for (int g = 0; g < 8; ++g) {
      int rl = wave * 16 + g + 8 * hb;      // row in 128-row tile = r*8 + j
      int r = rl >> 3, j = rl & 7;
      int n = n0 + r;
      int orow = (b * 8 + (n >> 8)) * 2048 + ((n & 255) << 3) + j;
      out[(size_t)orow * 128 + ct * 16 + (lane & 15)] = ao[g] + bov;
    }
  }
}

// ---------------------------------------------------------------------------
extern "C" void kernel_launch(void* const* d_in, const int* in_sizes, int n_in,
                              void* d_out, int out_size, void* d_ws,
                              size_t ws_size, hipStream_t stream) {
  (void)in_sizes; (void)n_in; (void)out_size; (void)ws_size;
  // setup_inputs order: x_0, edge_attr, q_data, Wq,bq, Wk0,bk0, Wv0,bv0,
  //                     Wk1,bk1, Wv1,bv1, Wo,bo.  (x_0 branch is dead.)
  const float* edge = (const float*)d_in[1];
  const float* qdat = (const float*)d_in[2];
  const float* Wq   = (const float*)d_in[3];
  const float* bq   = (const float*)d_in[4];
  const float* Wk1  = (const float*)d_in[9];
  const float* bk1  = (const float*)d_in[10];
  const float* Wv1  = (const float*)d_in[11];
  const float* bv1  = (const float*)d_in[12];
  const float* Wo   = (const float*)d_in[13];
  const float* bo   = (const float*)d_in[14];

  char*  ws       = (char*)d_ws;
  float* kv_acc   = (float*)ws;                 // 32*16*16 f32 = 32768 B
  float* ksum_acc = (float*)(ws + 32768);       // 32*16 f32    = 2048 B
  bf16*  WkBf     = (bf16*)(ws + 34816);        // 128*128 bf16 each
  bf16*  WvBf     = (bf16*)(ws + 67584);
  bf16*  WqBf     = (bf16*)(ws + 100352);
  bf16*  WoBf     = (bf16*)(ws + 133120);       // total 165888 B

  prep_kernel<<<256, 256, 0, stream>>>(Wk1, Wv1, Wq, Wo, WkBf, WvBf, WqBf, WoBf,
                                       kv_acc);
  kv_reduce_kernel<<<dim3(8, 32), 256, 176640, stream>>>(edge, bk1, bv1, WkBf,
                                                         WvBf, kv_acc, ksum_acc);
  out_kernel<<<dim3(128, 4), 256, 151552, stream>>>(qdat, bq, bo, WqBf, WoBf,
                                                    kv_acc, ksum_acc,
                                                    (float*)d_out);
}
